// GraphFireDetector_57260503990594
// MI455X (gfx1250) — matrix operand order repaired and checked by reading, not compiled, not run on hardware
//
#include <hip/hip_runtime.h>
#include <hip/hip_fp16.h>

typedef __attribute__((ext_vector_type(16))) _Float16 v16h;
typedef __attribute__((ext_vector_type(8)))  _Float16 v8h;
typedef __attribute__((ext_vector_type(8)))  float    v8f;

#define HD   128
#define SEQ  2048
#define NBAT 32
#define NROWS (NBAT * SEQ)

// ---------------------------------------------------------------------------
// Swizzle a [128][128] f32 weight matrix into WMMA B-fragment order (f16).
// Fragment (nt,kt): lane l, element e -> K = 32*kt + 16*(l>>4) + e,
//                                        N = 16*nt + (l&15)
// dst flat index: ((nt*4+kt)*32 + l)*16 + e   (32 contiguous bytes per lane)
// ---------------------------------------------------------------------------
__global__ void swizzle_w_kernel(const float* __restrict__ w,
                                 _Float16* __restrict__ dst) {
    int t = threadIdx.x;
    for (int i = t; i < 128 * 128; i += 256) {
        int e    = i & 15;
        int l    = (i >> 4) & 31;
        int frag = i >> 9;            // nt*4 + kt
        int kt   = frag & 3;
        int nt   = frag >> 2;
        int K    = 32 * kt + 16 * (l >> 4) + e;
        int N    = 16 * nt + (l & 15);
        dst[i]   = (_Float16)w[K * 128 + N];
    }
}

__global__ void zero_kernel(float* p, int n) {
    int i = blockIdx.x * 256 + threadIdx.x;
    if (i < n) p[i] = 0.0f;
}

// ---------------------------------------------------------------------------
// Encoder layer 1: t = relu(x @ enc_w1 + enc_b1), K=6 -> plain VALU.
// One thread per row.
// ---------------------------------------------------------------------------
__global__ __launch_bounds__(256)
void enc1_kernel(const float* __restrict__ x, const float* __restrict__ w1,
                 const float* __restrict__ b1, _Float16* __restrict__ out) {
    __shared__ float sw[6 * 128];
    __shared__ float sb[128];
    int t = threadIdx.x;
    for (int i = t; i < 6 * 128; i += 256) sw[i] = w1[i];
    if (t < 128) sb[t] = b1[t];
    __syncthreads();

    int r = blockIdx.x * 256 + t;
    float xv[6];
#pragma unroll
    for (int k = 0; k < 6; k++) xv[k] = x[(size_t)r * 6 + k];
    for (int n = 0; n < 128; n++) {
        float a = sb[n];
#pragma unroll
        for (int k = 0; k < 6; k++) a += xv[k] * sw[k * 128 + n];
        a = fmaxf(a, 0.0f);
        out[(size_t)r * 128 + n] = (_Float16)a;
    }
}

// ---------------------------------------------------------------------------
// Generic layer: z = in @ W + b  (WMMA f16 -> f32 accum), then optionally
// degree-normalized tridiagonal aggregate along seq, optional relu,
// store f16, optional pooled accumulation.
// Block: 64 seq rows x 128 cols, 8 waves = (4 row tiles) x (2 col halves).
// ---------------------------------------------------------------------------
__global__ __launch_bounds__(256)
void gemm_layer_kernel(const _Float16* __restrict__ in,     // [NROWS][128] f16
                       const _Float16* __restrict__ wswz,   // swizzled f16 W
                       const float* __restrict__ wfull,     // [128][128] f32
                       const float* __restrict__ bias,      // [128]
                       _Float16* __restrict__ out,          // [NROWS][128] f16
                       float* pooled,                       // [NBAT][128] or null
                       int do_agg, int do_relu) {
    __shared__ float zbuf[66 * 128];    // rows -1..64 at index r+1

    const int t    = threadIdx.x;
    const int lane = t & 31;
    const int w    = t >> 5;
    const int b    = blockIdx.x >> 5;       // batch
    const int s0   = (blockIdx.x & 31) * 64; // seq chunk start
    const int rowbase = b * SEQ + s0;
    const int rt = w & 3;                   // row tile 0..3
    const int nh = w >> 2;                  // col half 0..1
    const int g  = lane >> 4;
    const int lm = lane & 15;

    // ---- load A fragments (kept in VGPRs for all 4 k-steps) ----
    const _Float16* arow = in + (size_t)(rowbase + rt * 16 + lm) * 128;
    v16h afrag[4];
#pragma unroll
    for (int kt = 0; kt < 4; kt++) {
        v8h lo = *(const v8h*)(arow + 32 * kt + 16 * g);
        v8h hi = *(const v8h*)(arow + 32 * kt + 16 * g + 8);
#pragma unroll
        for (int e = 0; e < 8; e++) { afrag[kt][e] = lo[e]; afrag[kt][e + 8] = hi[e]; }
    }

    // ---- 4 output tiles per wave: acc = A @ B, + bias, -> LDS ----
#pragma unroll
    for (int nti = 0; nti < 4; nti++) {
        int nt  = nh * 4 + nti;
        int col = nt * 16 + lm;
        float bv = bias[col];
        v8f acc = {};
#pragma unroll
        for (int kt = 0; kt < 4; kt++) {
            v16h bfrag = *(const v16h*)(wswz + (nt * 4 + kt) * 512 + lane * 16);
            acc = __builtin_amdgcn_wmma_f32_16x16x32_f16(
                false, afrag[kt], false, bfrag, (short)0, acc, false, false);
        }
#pragma unroll
        for (int r = 0; r < 8; r++)
            zbuf[(1 + rt * 16 + r + 8 * g) * 128 + col] = acc[r] + bv;
    }

    // ---- halo rows (seq s0-1 and s0+64) via VALU dot products ----
    if (do_agg) {
        int hrow = t >> 7;              // 0 -> row -1, 1 -> row 64
        int col  = t & 127;
        int s    = hrow ? (s0 + 64) : (s0 - 1);
        float z  = 0.0f;
        if (s >= 0 && s < SEQ) {
            const _Float16* hr = in + (size_t)(b * SEQ + s) * 128;
            float a = bias[col];
            for (int k = 0; k < 128; k++) a += (float)hr[k] * wfull[k * 128 + col];
            z = a;
        }
        zbuf[(hrow ? 65 : 0) * 128 + col] = z;
    }
    __syncthreads();

    // ---- epilogue: aggregate / relu / store / pool ----
    for (int i = t; i < 64 * 128; i += 256) {
        int r = i >> 7;
        int c = i & 127;
        float v;
        if (do_agg) {
            int s = s0 + r;
            float sum = zbuf[r * 128 + c] + zbuf[(r + 1) * 128 + c]
                      + zbuf[(r + 2) * 128 + c];
            float deg = (s == 0 || s == SEQ - 1) ? 2.0f : 3.0f;
            v = sum / (deg + 1e-8f);
        } else {
            v = zbuf[(r + 1) * 128 + c];
        }
        if (do_relu) v = fmaxf(v, 0.0f);
        out[(size_t)(rowbase + r) * 128 + c] = (_Float16)v;
        if (pooled) atomicAdd(&pooled[b * 128 + c], v);
    }
}

// ---------------------------------------------------------------------------
// Classifier: out = relu(mean @ cls_w1 + cls_b1) @ cls_w2 + cls_b2
// ---------------------------------------------------------------------------
__global__ __launch_bounds__(256)
void classifier_kernel(const float* __restrict__ pooled,
                       const float* __restrict__ cw1, const float* __restrict__ cb1,
                       const float* __restrict__ cw2, const float* __restrict__ cb2,
                       float* __restrict__ out) {
    __shared__ float sp[32 * 128];
    __shared__ float st[32 * 64];
    int t = threadIdx.x;
    for (int i = t; i < 32 * 128; i += 256) sp[i] = pooled[i] * (1.0f / SEQ);
    __syncthreads();
    for (int i = t; i < 32 * 64; i += 256) {
        int bi = i >> 6, n = i & 63;
        float a = cb1[n];
        for (int k = 0; k < 128; k++) a += sp[bi * 128 + k] * cw1[k * 64 + n];
        st[i] = fmaxf(a, 0.0f);
    }
    __syncthreads();
    if (t < 96) {
        int bi = t / 3, n = t % 3;
        float a = cb2[n];
        for (int k = 0; k < 64; k++) a += st[bi * 64 + k] * cw2[k * 3 + n];
        out[bi * 3 + n] = a;
    }
}

// ---------------------------------------------------------------------------
extern "C" void kernel_launch(void* const* d_in, const int* in_sizes, int n_in,
                              void* d_out, int out_size, void* d_ws, size_t ws_size,
                              hipStream_t stream) {
    (void)in_sizes; (void)n_in; (void)out_size; (void)ws_size;
    const float* x      = (const float*)d_in[0];
    const float* enc_w1 = (const float*)d_in[1];
    const float* enc_b1 = (const float*)d_in[2];
    const float* enc_w2 = (const float*)d_in[3];
    const float* enc_b2 = (const float*)d_in[4];
    const float* gc1_w  = (const float*)d_in[5];
    const float* gc1_b  = (const float*)d_in[6];
    const float* gc2_w  = (const float*)d_in[7];
    const float* gc2_b  = (const float*)d_in[8];
    const float* gc3_w  = (const float*)d_in[9];
    const float* gc3_b  = (const float*)d_in[10];
    const float* cls_w1 = (const float*)d_in[11];
    const float* cls_b1 = (const float*)d_in[12];
    const float* cls_w2 = (const float*)d_in[13];
    const float* cls_b2 = (const float*)d_in[14];

    char* ws = (char*)d_ws;
    const size_t BUF_BYTES = (size_t)NROWS * 128 * sizeof(_Float16); // 16 MB
    _Float16* buf0 = (_Float16*)ws;
    _Float16* buf1 = (_Float16*)(ws + BUF_BYTES);
    _Float16* wsw  = (_Float16*)(ws + 2 * BUF_BYTES);                // 4 x 32 KB
    float*  pooled = (float*)(ws + 2 * BUF_BYTES + 4 * 16384 * sizeof(_Float16));

    // one-time-per-call weight swizzles (deterministic, on stream)
    swizzle_w_kernel<<<1, 256, 0, stream>>>(enc_w2, wsw + 0 * 16384);
    swizzle_w_kernel<<<1, 256, 0, stream>>>(gc1_w,  wsw + 1 * 16384);
    swizzle_w_kernel<<<1, 256, 0, stream>>>(gc2_w,  wsw + 2 * 16384);
    swizzle_w_kernel<<<1, 256, 0, stream>>>(gc3_w,  wsw + 3 * 16384);
    zero_kernel<<<(NBAT * 128 + 255) / 256, 256, 0, stream>>>(pooled, NBAT * 128);

    // encoder layer 1 (K=6, VALU)
    enc1_kernel<<<NROWS / 256, 256, 0, stream>>>(x, enc_w1, enc_b1, buf0);

    // encoder layer 2: no agg, no relu
    gemm_layer_kernel<<<1024, 256, 0, stream>>>(buf0, wsw + 0 * 16384, enc_w2,
                                                enc_b2, buf1, nullptr, 0, 0);
    // three GCN layers: agg + relu; last one also pools
    gemm_layer_kernel<<<1024, 256, 0, stream>>>(buf1, wsw + 1 * 16384, gc1_w,
                                                gc1_b, buf0, nullptr, 1, 1);
    gemm_layer_kernel<<<1024, 256, 0, stream>>>(buf0, wsw + 2 * 16384, gc2_w,
                                                gc2_b, buf1, nullptr, 1, 1);
    gemm_layer_kernel<<<1024, 256, 0, stream>>>(buf1, wsw + 3 * 16384, gc3_w,
                                                gc3_b, buf0, pooled, 1, 1);

    classifier_kernel<<<1, 256, 0, stream>>>(pooled, cls_w1, cls_b1, cls_w2,
                                             cls_b2, (float*)d_out);
}
